// SeparableConv2d_deform_58394375356459
// MI455X (gfx1250) — matrix operand order repaired and checked by reading, not compiled, not run on hardware
//
#include <hip/hip_runtime.h>
#include <hip/hip_bf16.h>
#include <stddef.h>

// ---------------- problem constants ----------------
#define BN   8      // batch
#define CC   256    // channels
#define HH   96
#define WW   96
#define HWO  (HH*WW)        // 9216
#define HP   98             // padded (hp, wp)
#define HW98 (HP*HP)        // 9604
#define OCO  512            // offset-conv out channels (2C)
#define POUT 256
#define XPH  100            // conv staging height (pad 2)
#define XPW  120            // conv staging width  (pad 2 + tile tail, zero filled)

// ---------------- workspace layout (bytes, all 256-aligned) ----------------
#define OFF_WA    ((size_t)0)                                   // bf16 [9][512][256]
#define SZ_WA     ((size_t)9*OCO*CC*2)                          // 2,359,296
#define OFF_WPW   (OFF_WA + SZ_WA)                              // bf16 [256][256]
#define SZ_WPW    ((size_t)POUT*CC*2)
#define OFF_XPBT  (OFF_WPW + SZ_WPW)                            // bf16 [B][100][120][256]
#define SZ_XPBT   ((size_t)BN*XPH*XPW*CC*2)                     // 49,152,000
#define OFF_OFFS  (OFF_XPBT + SZ_XPBT)                          // f32 [B][512][98][98]; later reused as xrT bf16 [B][9216][256]
#define SZ_OFFS   ((size_t)BN*OCO*HW98*4)                       // 157,351,936 (>= xrT's 37.7MB)
#define OFF_XD    (OFF_OFFS + SZ_OFFS)                          // f32 [B][256][9216]
#define SZ_XD     ((size_t)BN*CC*HWO*4)
#define OFF_STATS (OFF_XD + SZ_XD)                              // f32 sum[256], sumsq[256], scale[256], shift[256]

// ---------------- CDNA5 async-to-LDS availability (probe via __has_builtin) ----
#if defined(__has_builtin)
# if __has_builtin(__builtin_amdgcn_global_load_async_to_lds_b128)
#  define HAVE_ASYNC_LDS 1
# endif
#endif
#ifndef HAVE_ASYNC_LDS
# define HAVE_ASYNC_LDS 0
#endif

#define AS1 __attribute__((address_space(1)))
#define AS3 __attribute__((address_space(3)))

typedef __attribute__((ext_vector_type(4)))  int    v4i;

__device__ __forceinline__ void wait_async_zero() {
#if HAVE_ASYNC_LDS
# if __has_builtin(__builtin_amdgcn_s_wait_asynccnt)
  __builtin_amdgcn_s_wait_asynccnt(0);
# else
  asm volatile("s_wait_asynccnt 0" ::: "memory");
# endif
#endif
}

// ---------------- WMMA fragment types ----------------
typedef __attribute__((ext_vector_type(16))) __bf16 bf16x16;
typedef __attribute__((ext_vector_type(8)))  __bf16 bf16x8;
typedef __attribute__((ext_vector_type(8)))  float  f32x8;

// A 16x32 bf16 fragment (ISA 7.12.2): lane L holds row M=L%16; half=L/16 selects
// K runs [8*half, 8*half+8) (VGPR0-3) and [16+8*half, +8) (VGPR4-7).
__device__ __forceinline__ bf16x16 load_a_frag(const __bf16* base, int hf) {
  bf16x8 lo = *reinterpret_cast<const bf16x8*>(base + 8*hf);
  bf16x8 hi = *reinterpret_cast<const bf16x8*>(base + 8*hf + 16);
  bf16x16 a;
#pragma unroll
  for (int i = 0; i < 8; ++i) { a[i] = lo[i]; a[i+8] = hi[i]; }
  return a;
}

// B 32x16 bf16 fragment: lane L holds column N=L%16; half selects K runs
// [8*half, +8) and [16+8*half, +8). 'base' points at the K-contiguous column.
__device__ __forceinline__ bf16x16 load_b_frag(const __bf16* base, int hf) {
  bf16x8 lo = *reinterpret_cast<const bf16x8*>(base + 8*hf);
  bf16x8 hi = *reinterpret_cast<const bf16x8*>(base + 8*hf + 16);
  bf16x16 bv;
#pragma unroll
  for (int i = 0; i < 8; ++i) { bv[i] = lo[i]; bv[i+8] = hi[i]; }
  return bv;
}

// ---------------- 1) weight prep ----------------
__global__ void k_prep_weights(const float* __restrict__ w_off,
                               const float* __restrict__ w_pw,
                               __bf16* __restrict__ wA,
                               __bf16* __restrict__ wpwb) {
  int i = blockIdx.x * blockDim.x + threadIdx.x;
  const int nA = 9 * OCO * CC;
  if (i < nA) {
    int kpos = i / (OCO * CC);
    int r    = i - kpos * (OCO * CC);
    int oc = r >> 8, ic = r & 255;
    wA[i] = (__bf16)w_off[(size_t)(oc * CC + ic) * 9 + kpos];
  }
  if (i < POUT * CC) wpwb[i] = (__bf16)w_pw[i];
}

// ---------------- 2) input prep: zero-padded NHWC bf16 staging ----------------
__global__ void k_prep_input(const float* __restrict__ x, __bf16* __restrict__ xpbT) {
  size_t i = (size_t)blockIdx.x * blockDim.x + threadIdx.x;
  const size_t total = (size_t)BN * XPH * XPW * CC;
  if (i >= total) return;
  int ic = (int)(i & 255);
  size_t t = i >> 8;
  int xx = (int)(t % XPW); t /= XPW;
  int yy = (int)(t % XPH);
  int b  = (int)(t / XPH);
  int ys = yy - 2, xs = xx - 2;
  float v = 0.f;
  if (ys >= 0 && ys < HH && xs >= 0 && xs < WW)
    v = x[(((size_t)b * CC + ic) * HH + ys) * WW + xs];
  xpbT[i] = (__bf16)v;
}

// ---------------- 3) offset conv as implicit GEMM on WMMA ----------------
// Block = 128 threads = 4 waves; block tile = 256 oc x 16 x (wave = 64 oc x 16 x,
// 4 accumulators).  The B tile (32k x 16n bf16, 1KB) is identical for all waves:
// wave 0 stages it into LDS via GLOBAL_LOAD_ASYNC_TO_LDS_B128 (ASYNCcnt), then a
// workgroup barrier publishes it; all waves consume via ds_load B-fragments.
// grid = (7 x-tiles, 98 rows, 8 batches * 2 oc-groups).
__global__ void __launch_bounds__(128)
k_conv_offsets(const __bf16* __restrict__ wA,
               const __bf16* __restrict__ xpbT,
               float* __restrict__ offs) {
  __shared__ __bf16 Btile[16][72];                  // [n][k], padded rows (144B)
  const int lane = threadIdx.x & 31;
  const int wave = threadIdx.x >> 5;
  const int hf   = lane >> 4;
  const int l16  = lane & 15;
  const int x0   = blockIdx.x * 16;
  const int y    = blockIdx.y;
  const int b    = blockIdx.z >> 1;
  const int oc0  = (blockIdx.z & 1) * 256 + wave * 64;

  const __bf16* xb = xpbT + (size_t)b * (XPH * XPW * CC);

  f32x8 acc[4] = {};
  for (int kpos = 0; kpos < 9; ++kpos) {
    const int ky = kpos / 3, kx = kpos - 3 * ky;
    const __bf16* wrow = wA + (size_t)kpos * OCO * CC;
    for (int kc = 0; kc < CC; kc += 32) {
      if (wave == 0) {
        // stage 16 rows x 64B: 64 16-byte chunks across 32 lanes x 2 issues
#pragma unroll
        for (int it = 0; it < 2; ++it) {
          int c = lane + it * 32;
          int n = c >> 2, part = c & 3;
          const __bf16* src =
              xb + ((size_t)(y + ky) * XPW + (x0 + n + kx)) * CC + kc + part * 8;
          __bf16* dst = &Btile[n][part * 8];
#if HAVE_ASYNC_LDS
          __builtin_amdgcn_global_load_async_to_lds_b128(
              (AS1 v4i*)src, (AS3 v4i*)dst, 0, 0);
#else
          *reinterpret_cast<bf16x8*>(dst) =
              *reinterpret_cast<const bf16x8*>(src);
#endif
        }
        __builtin_prefetch(xb + ((size_t)(y + ky) * XPW + (x0 + lane + kx)) * CC
                               + kc + 32, 0, 1);    // global_prefetch_b8 next chunk
        wait_async_zero();
      }
      __syncthreads();                              // tile visible block-wide
      bf16x16 bfr = load_b_frag(&Btile[l16][0], hf);
#pragma unroll
      for (int t = 0; t < 4; ++t) {
        bf16x16 af = load_a_frag(wrow + (size_t)(oc0 + 16 * t + l16) * CC + kc, hf);
        acc[t] = __builtin_amdgcn_wmma_f32_16x16x32_bf16(
            false, af, false, bfr, (short)0, acc[t], false, false);
      }
      __syncthreads();                              // done reading before re-stage
    }
  }
  const int x = x0 + l16;
  if (x < HP) {                                     // mask tail tile (112 -> 98)
#pragma unroll
    for (int t = 0; t < 4; ++t)
#pragma unroll
      for (int g = 0; g < 8; ++g) {
        int o = oc0 + 16 * t + hf * 8 + g;          // C/D layout: M = 8*half + g
        offs[(((size_t)b * OCO + o) * HP + y) * HP + x] = acc[t][g];
      }
  }
}

// ---------------- 4) deformable bilinear sampling + BN statistics ----------------
__device__ __forceinline__ float samp_pad1(const float* m, int r, int c) {
  return (r >= 1 && r <= HH && c >= 1 && c <= WW) ? m[(r - 1) * WW + (c - 1)] : 0.f;
}

__global__ void __launch_bounds__(256)
k_sample_stats(const float* __restrict__ x,
               const float* __restrict__ offs,
               float* __restrict__ xd,
               float* __restrict__ sum, float* __restrict__ sumsq) {
  const int bc = blockIdx.x;            // b*256 + j
  const int b  = bc >> 8;
  const int j  = bc & 255;
  const float* xmap = x + (size_t)bc * HWO;
  // flat-reshape quirk: off_y(j,p) = offs_flat[b][2j*HW98 + 2p], off_x at +1
  const float* op = offs + ((size_t)b * OCO + 2 * j) * (size_t)HW98;
  float* xdp = xd + (size_t)bc * HWO;

  float ls = 0.f, lq = 0.f;
  for (int p = threadIdx.x; p < HWO; p += 256) {
    int yo = p / WW, xo = p - yo * WW;
    int yy = yo + 1, xx = xo + 1;                    // crop [1:-1,1:-1]
    int pp = yy * HP + xx;
    float cy = op[2 * pp]     + (float)yy;
    float cx = op[2 * pp + 1] + (float)xx;
    cy = fminf(fmaxf(cy, 0.f), (float)(HP - 1));
    cx = fminf(fmaxf(cx, 0.f), (float)(HP - 1));
    float y0f = floorf(cy), x0f = floorf(cx);
    int y0 = (int)y0f, xq = (int)x0f;
    int y1 = min(y0 + 1, HP - 1), x1 = min(xq + 1, HP - 1);
    float dy = cy - y0f, dx = cx - x0f;
    float v00 = samp_pad1(xmap, y0, xq);
    float v01 = samp_pad1(xmap, y0, x1);
    float v10 = samp_pad1(xmap, y1, xq);
    float v11 = samp_pad1(xmap, y1, x1);
    float top = v00 + (v01 - v00) * dx;
    float bot = v10 + (v11 - v10) * dx;
    float v   = top + (bot - top) * dy;
    xdp[p] = v;
    ls += v; lq += v * v;
  }
  __shared__ float ss[256], sq[256];
  ss[threadIdx.x] = ls; sq[threadIdx.x] = lq;
  __syncthreads();
  for (int s = 128; s > 0; s >>= 1) {
    if (threadIdx.x < s) {
      ss[threadIdx.x] += ss[threadIdx.x + s];
      sq[threadIdx.x] += sq[threadIdx.x + s];
    }
    __syncthreads();
  }
  if (threadIdx.x == 0) {
    atomicAdd(&sum[j],   ss[0]);
    atomicAdd(&sumsq[j], sq[0]);
  }
}

// ---------------- 5) fold BN stats into per-channel scale/shift ----------------
__global__ void k_finalize_stats(const float* __restrict__ sum,
                                 const float* __restrict__ sumsq,
                                 const float* __restrict__ gamma,
                                 const float* __restrict__ beta,
                                 float* __restrict__ scale,
                                 float* __restrict__ shift) {
  int c = threadIdx.x;
  const float N = (float)BN * (float)HWO;
  float m  = sum[c] / N;
  float va = sumsq[c] / N - m * m;                   // biased variance
  float is = rsqrtf(va + 1e-5f);
  float sc = gamma[c] * is;
  scale[c] = sc;
  shift[c] = beta[c] - m * sc;
}

// ---------------- 6) normalize + ReLU + bf16 + transpose to [hw][c] ----------------
__global__ void __launch_bounds__(256)
k_norm_transpose(const float* __restrict__ xd,
                 const float* __restrict__ scale,
                 const float* __restrict__ shift,
                 __bf16* __restrict__ xrT) {
  __shared__ __bf16 tile[16][17];
  const int hw0 = blockIdx.x * 16;
  const int c0  = blockIdx.y * 16;
  const int b   = blockIdx.z;
  int r = threadIdx.x >> 4, s = threadIdx.x & 15;
  int c = c0 + r;
  float v = xd[((size_t)b * CC + c) * HWO + hw0 + s];
  v = fmaxf(v * scale[c] + shift[c], 0.f);
  tile[r][s] = (__bf16)v;
  __syncthreads();
  xrT[((size_t)b * HWO + hw0 + r) * CC + c0 + s] = tile[s][r];
}

// ---------------- 7) pointwise 1x1 conv as WMMA GEMM ----------------
// Block = 128 threads = 4 waves; block covers all 256 o x 16 hw.
// Wave tile = 64 o x 16 hw: one B-frag reused by 4 A-frags/wmma per K-chunk.
__global__ void __launch_bounds__(128)
k_pointwise(const __bf16* __restrict__ wpwb,
            const __bf16* __restrict__ xrT,
            float* __restrict__ out) {
  const int lane = threadIdx.x & 31;
  const int wave = threadIdx.x >> 5;
  const int hf   = lane >> 4;
  const int l16  = lane & 15;
  const int hw0  = blockIdx.x * 16;
  const int o0   = wave * 64;
  const int b    = blockIdx.y;

  const __bf16* bcol = xrT + ((size_t)b * HWO + hw0 + l16) * CC;

  f32x8 acc[4] = {};
#pragma unroll
  for (int kc = 0; kc < CC; kc += 32) {
    bf16x16 bfr = load_b_frag(bcol + kc, hf);
#pragma unroll
    for (int t = 0; t < 4; ++t) {
      bf16x16 af = load_a_frag(wpwb + (size_t)(o0 + 16 * t + l16) * CC + kc, hf);
      acc[t] = __builtin_amdgcn_wmma_f32_16x16x32_bf16(
          false, af, false, bfr, (short)0, acc[t], false, false);
    }
  }
#pragma unroll
  for (int t = 0; t < 4; ++t)
#pragma unroll
    for (int g = 0; g < 8; ++g) {
      int o = o0 + 16 * t + hf * 8 + g;
      out[((size_t)b * POUT + o) * HWO + hw0 + l16] = acc[t][g];
    }
}

// ---------------- launch ----------------
extern "C" void kernel_launch(void* const* d_in, const int* in_sizes, int n_in,
                              void* d_out, int out_size, void* d_ws, size_t ws_size,
                              hipStream_t stream) {
  const float* x     = (const float*)d_in[0];   // (8,256,96,96)
  const float* w_off = (const float*)d_in[1];   // (512,256,3,3)
  const float* gamma = (const float*)d_in[2];   // (256,)
  const float* beta  = (const float*)d_in[3];   // (256,)
  const float* w_pw  = (const float*)d_in[4];   // (256,256)
  float* out = (float*)d_out;                   // (8,256,96,96)

  char* ws = (char*)d_ws;
  __bf16* wA    = (__bf16*)(ws + OFF_WA);
  __bf16* wpwb  = (__bf16*)(ws + OFF_WPW);
  __bf16* xpbT  = (__bf16*)(ws + OFF_XPBT);
  float*  offs  = (float*)(ws + OFF_OFFS);
  __bf16* xrT   = (__bf16*)(ws + OFF_OFFS);     // reuse: offsets dead after sampling
  float*  xd    = (float*)(ws + OFF_XD);
  float*  sum   = (float*)(ws + OFF_STATS);
  float*  sumsq = sum + 256;
  float*  scale = sum + 512;
  float*  shift = sum + 768;

  (void)hipMemsetAsync(sum, 0, 2 * 256 * sizeof(float), stream);

  {
    int n = 9 * OCO * CC;
    k_prep_weights<<<(n + 255) / 256, 256, 0, stream>>>(w_off, w_pw, wA, wpwb);
  }
  {
    size_t n = (size_t)BN * XPH * XPW * CC;
    k_prep_input<<<(unsigned)((n + 255) / 256), 256, 0, stream>>>(x, xpbT);
  }
  // offset conv (WMMA): grid (7 x-tiles, 98 rows, 8 b * 2 oc-groups)
  k_conv_offsets<<<dim3(7, HP, BN * 2), 128, 0, stream>>>(wA, xpbT, offs);
  // deformable sampling + BN stats
  k_sample_stats<<<BN * CC, 256, 0, stream>>>(x, offs, xd, sum, sumsq);
  k_finalize_stats<<<1, 256, 0, stream>>>(sum, sumsq, gamma, beta, scale, shift);
  k_norm_transpose<<<dim3(HWO / 16, CC / 16, BN), 256, 0, stream>>>(xd, scale, shift, xrT);
  // pointwise 1x1 conv (WMMA): grid (576 hw-tiles, 8 b)
  k_pointwise<<<dim3(HWO / 16, BN), 128, 0, stream>>>(wpwb, xrT, out);
}
